// DistanceWeightedSelfAttention_84327387890256
// MI455X (gfx1250) — matrix operand order repaired and checked by reading, not compiled, not run on hardware
//
#include <hip/hip_runtime.h>
#include <math.h>

typedef __bf16 bf16_t;
typedef __attribute__((ext_vector_type(16))) __bf16 v16bf;
typedef __attribute__((ext_vector_type(8)))  __bf16 v8bf;
typedef __attribute__((ext_vector_type(8)))  float  v8f;
typedef __attribute__((ext_vector_type(4)))  int    v4i;

#define LAMBDA_DECAY 0.1f
#define B_  8
#define N_  2048
#define D_  128
#define SCALE_ 0.08838834764831845f   // 1/sqrt(128)

#if defined(__HIP_DEVICE_COMPILE__) && __has_builtin(__builtin_amdgcn_global_load_async_to_lds_b128)
#define HAS_ASYNC_LDS 1
#else
#define HAS_ASYNC_LDS 0
#endif

// ---------------- WMMA helpers (CDNA5 16x16x32 bf16, wave32) ----------------

__device__ __forceinline__ v8f zero8() {
  v8f z;
#pragma unroll
  for (int i = 0; i < 8; ++i) z[i] = 0.0f;
  return z;
}

// A-fragment: 16x32 tile, row-major memory (rows m, contiguous k), stride ld.
// Lanes 0-15 -> M=lane; elems e -> K = 16*half + e (matches load_b_frag).
__device__ __forceinline__ v16bf load_a_frag(const bf16_t* base, int ld) {
  const int lane = threadIdx.x & 31;
  const int half = lane >> 4;
  const int m    = lane & 15;
  const bf16_t* row = base + (size_t)m * ld;
  v8bf lo = *(const v8bf*)(row + half * 16);
  v8bf hi = *(const v8bf*)(row + half * 16 + 8);
  return __builtin_shufflevector(lo, hi, 0,1,2,3,4,5,6,7,8,9,10,11,12,13,14,15);
}

// B-fragment: 32x16 tile from N-major memory (rows n, contiguous k), stride ld.
// lane -> N=lane&15, elems e -> K = 16*half + e (one contiguous 32B read).
__device__ __forceinline__ v16bf load_b_frag(const bf16_t* base, int ld) {
  const int lane = threadIdx.x & 31;
  const int half = lane >> 4;
  const int n    = lane & 15;
  return *(const v16bf*)(base + (size_t)n * ld + half * 16);
}

__device__ __forceinline__ v8f wmma_bf16(v16bf a, v16bf b, v8f c) {
  return __builtin_amdgcn_wmma_f32_16x16x32_bf16(
      false, a, false, b, (short)0, c, false, false);
}

// ---------------- async Global -> LDS copy (16B granularity) ----------------

#if HAS_ASYNC_LDS
typedef v4i __attribute__((address_space(1))) *gvp_t;   // global int4*
typedef v4i __attribute__((address_space(3))) *lvp_t;   // LDS int4*
#endif

__device__ __forceinline__ void async_cp16(bf16_t* lds_dst, const bf16_t* gsrc) {
#if HAS_ASYNC_LDS
  __builtin_amdgcn_global_load_async_to_lds_b128((gvp_t)gsrc, (lvp_t)lds_dst, 0, 0);
#else
  *(v8bf*)lds_dst = *(const v8bf*)gsrc;     // synchronous fallback (ds_store)
#endif
}

__device__ __forceinline__ void wait_async() {
#if HAS_ASYNC_LDS
#if __has_builtin(__builtin_amdgcn_s_wait_asynccnt)
  __builtin_amdgcn_s_wait_asynccnt(0);
#else
  asm volatile("s_wait_asynccnt 0x0" ::: "memory");
#endif
#endif
}

// ---------------- conversion kernels ----------------

__global__ void k_cvt_x(const float* __restrict__ X, bf16_t* __restrict__ Xbf, int n) {
  int i = blockIdx.x * blockDim.x + threadIdx.x;
  if (i < n) Xbf[i] = (bf16_t)X[i];
}

// 128x128 weight -> bf16, transposed (N-major): WT[n*128+k] = W[k*128+n]
__global__ void k_cvt_wT(const float* __restrict__ W, bf16_t* __restrict__ WT) {
  for (int e = threadIdx.x; e < 128 * 128; e += blockDim.x) {
    int n = e >> 7, k = e & 127;
    WT[e] = (bf16_t)W[k * 128 + n];
  }
}

// ---------------- QKV projection: [16384,128] x [128,128] bf16 WMMA ----------------

__global__ __launch_bounds__(256)
void k_qkv(const bf16_t* __restrict__ Xbf,
           const bf16_t* __restrict__ WqT, const bf16_t* __restrict__ WkT,
           const bf16_t* __restrict__ WvT,
           bf16_t* __restrict__ Qbf, bf16_t* __restrict__ Kbf, bf16_t* __restrict__ Vt) {
  const int wave = threadIdx.x >> 5;
  const int lane = threadIdx.x & 31;
  const int half = lane >> 4;
  const int nl   = lane & 15;
  const int m0   = blockIdx.x * 128 + wave * 16;
  const int proj = blockIdx.y;                        // 0=Q 1=K 2=V
  const bf16_t* WT = (proj == 0) ? WqT : (proj == 1) ? WkT : WvT;

  v8f acc[8];
#pragma unroll
  for (int t = 0; t < 8; ++t) acc[t] = zero8();

#pragma unroll
  for (int kk = 0; kk < 4; ++kk) {
    v16bf a = load_a_frag(Xbf + (size_t)m0 * D_ + kk * 32, D_);
#pragma unroll
    for (int nt = 0; nt < 8; ++nt) {
      v16bf b = load_b_frag(WT + nt * 16 * D_ + kk * 32, D_);
      acc[nt] = wmma_bf16(a, b, acc[nt]);
    }
  }

  if (proj == 2) {
#pragma unroll
    for (int nt = 0; nt < 8; ++nt)
#pragma unroll
      for (int r = 0; r < 8; ++r) {
        int g   = m0 + r + 8 * half;
        int bb  = g >> 11;
        int tok = g & (N_ - 1);
        int d   = nt * 16 + nl;
        Vt[(size_t)bb * D_ * N_ + (size_t)d * N_ + tok] = (bf16_t)acc[nt][r];
      }
  } else {
    bf16_t* O = (proj == 0) ? Qbf : Kbf;
#pragma unroll
    for (int nt = 0; nt < 8; ++nt)
#pragma unroll
      for (int r = 0; r < 8; ++r) {
        int g = m0 + r + 8 * half;
        O[(size_t)g * D_ + nt * 16 + nl] = (bf16_t)acc[nt][r];
      }
  }
}

// ---------------- fused flash attention + output projection ----------------

__global__ __launch_bounds__(256)
void k_attn(const bf16_t* __restrict__ Qbf, const bf16_t* __restrict__ Kbf,
            const bf16_t* __restrict__ Vt, const float* __restrict__ sizes,
            const float* __restrict__ maskp, const bf16_t* __restrict__ WoT,
            float* __restrict__ out) {
  __shared__ __align__(32) bf16_t ldsK[128 * 128];    // K tile [key][d]   32 KB
  __shared__ __align__(32) bf16_t ldsV[128 * 128];    // V tile [d][key]   32 KB
  __shared__ __align__(32) bf16_t ldsP[8 * 16 * 128]; // per-wave P/ctx    32 KB
  __shared__ float ldsSq[128], ldsSk[128], ldsMk[128];

  const int wave = threadIdx.x >> 5;
  const int lane = threadIdx.x & 31;
  const int half = lane >> 4;
  const int nl   = lane & 15;
  const int b    = blockIdx.x >> 4;
  const int q0   = (blockIdx.x & 15) * 128;
  const size_t bN = (size_t)b * N_;
  const int qw   = q0 + wave * 16;

  if (threadIdx.x < 128) ldsSq[threadIdx.x] = sizes[bN + q0 + threadIdx.x];

  // Q fragments pinned in registers for the whole kernel
  v16bf qa[4];
#pragma unroll
  for (int kk = 0; kk < 4; ++kk)
    qa[kk] = load_a_frag(Qbf + (bN + qw) * D_ + kk * 32, D_);

  float rm[8], rl[8];
  v8f acc[8];
#pragma unroll
  for (int r = 0; r < 8; ++r) { rm[r] = -3.0e38f; rl[r] = 0.0f; }
#pragma unroll
  for (int t = 0; t < 8; ++t) acc[t] = zero8();

  bf16_t* myP = ldsP + wave * (16 * 128);
  const bf16_t* vtb = Vt + (size_t)b * D_ * N_;

  for (int kt = 0; kt < 16; ++kt) {
    const int k0 = kt * 128;
    __syncthreads();   // previous tile's LDS consumers done

    // ---- async-stage K tile (contiguous 32KB) and V tile (128 rows of 256B) ----
    {
      const bf16_t* kg = Kbf + (bN + k0) * D_;
#pragma unroll
      for (int i = 0; i < 8; ++i) {
        int c = threadIdx.x + i * 256;            // 16B-chunk index, 2048 chunks
        async_cp16(ldsK + c * 8, kg + c * 8);
      }
#pragma unroll
      for (int i = 0; i < 8; ++i) {
        int c = threadIdx.x + i * 256;
        int d = c >> 4;                           // 16 chunks per 128-elem row
        int o = (c & 15) * 8;
        async_cp16(ldsV + d * 128 + o, vtb + (size_t)d * N_ + k0 + o);
      }
    }
    if (threadIdx.x < 128) {
      ldsSk[threadIdx.x] = sizes[bN + k0 + threadIdx.x];
      ldsMk[threadIdx.x] = maskp[bN + k0 + threadIdx.x];
    }
    if (kt + 1 < 16)  // warm L2 for next tile (global_prefetch_b8)
      __builtin_prefetch(Kbf + (bN + k0 + 128) * D_ + (size_t)lane * 512, 0, 3);

    wait_async();      // this wave's async copies landed in LDS
    __syncthreads();   // ... and everyone else's too

    // ---- S = Q K^T (B-frags from LDS), scale + decay + mask in fp32 ----
    float L[8][8];
#pragma unroll
    for (int nt = 0; nt < 8; ++nt) {
      v8f c = zero8();
#pragma unroll
      for (int kk = 0; kk < 4; ++kk) {
        v16bf bfr = load_b_frag(ldsK + nt * 16 * 128 + kk * 32, 128);
        c = wmma_bf16(qa[kk], bfr, c);
      }
      float sj   = ldsSk[nt * 16 + nl];
      float addm = (1.0f - ldsMk[nt * 16 + nl]) * -1.0e9f;
#pragma unroll
      for (int r = 0; r < 8; ++r) {
        float si = ldsSq[wave * 16 + r + 8 * half];
        float w  = __expf(-LAMBDA_DECAY * fabsf(si - sj));
        L[nt][r] = c[r] * SCALE_ * w + addm;
      }
    }

    // ---- online softmax ----
#pragma unroll
    for (int r = 0; r < 8; ++r) {
      float m = rm[r];
#pragma unroll
      for (int nt = 0; nt < 8; ++nt) m = fmaxf(m, L[nt][r]);
#pragma unroll
      for (int off = 8; off >= 1; off >>= 1)
        m = fmaxf(m, __shfl_xor(m, off, 32));
      float corr = __expf(rm[r] - m);
      rm[r] = m;
      rl[r] *= corr;
#pragma unroll
      for (int nt = 0; nt < 8; ++nt) acc[nt][r] = acc[nt][r] * corr;
    }

    // ---- P = exp(L - m) -> wave-private LDS slice, accumulate row sums ----
#pragma unroll
    for (int r = 0; r < 8; ++r) {
      float s = 0.0f;
#pragma unroll
      for (int nt = 0; nt < 8; ++nt) {
        float p = __expf(L[nt][r] - rm[r]);
        s += p;
        myP[(r + 8 * half) * 128 + nt * 16 + nl] = (bf16_t)p;
      }
#pragma unroll
      for (int off = 8; off >= 1; off >>= 1)
        s += __shfl_xor(s, off, 32);
      rl[r] += s;
    }
    asm volatile("s_wait_dscnt 0" ::: "memory");   // wave-private LDS RAW
    __builtin_amdgcn_wave_barrier();

    // ---- ctx += P @ V (both operands from LDS) ----
#pragma unroll
    for (int kk = 0; kk < 4; ++kk) {
      v16bf pa = load_a_frag(myP + kk * 32, 128);
#pragma unroll
      for (int nt = 0; nt < 8; ++nt) {
        v16bf vb = load_b_frag(ldsV + nt * 16 * 128 + kk * 32, 128);
        acc[nt] = wmma_bf16(pa, vb, acc[nt]);
      }
    }
  }

  // ---- normalize, stage ctx as bf16, fused output projection ----
#pragma unroll
  for (int r = 0; r < 8; ++r) {
    float inv = 1.0f / rl[r];
#pragma unroll
    for (int nt = 0; nt < 8; ++nt)
      myP[(r + 8 * half) * 128 + nt * 16 + nl] = (bf16_t)(acc[nt][r] * inv);
  }
  asm volatile("s_wait_dscnt 0" ::: "memory");
  __builtin_amdgcn_wave_barrier();

  v8f oacc[8];
#pragma unroll
  for (int t = 0; t < 8; ++t) oacc[t] = zero8();
#pragma unroll
  for (int kk = 0; kk < 4; ++kk) {
    v16bf ca = load_a_frag(myP + kk * 32, 128);
#pragma unroll
    for (int nt = 0; nt < 8; ++nt) {
      v16bf wb = load_b_frag(WoT + nt * 16 * D_ + kk * 32, D_);
      oacc[nt] = wmma_bf16(ca, wb, oacc[nt]);
    }
  }

  float* orow = out + (bN + qw) * D_;
#pragma unroll
  for (int nt = 0; nt < 8; ++nt)
#pragma unroll
    for (int r = 0; r < 8; ++r)
      orow[(size_t)(r + 8 * half) * D_ + nt * 16 + nl] = oacc[nt][r];
}

// ---------------- launch ----------------

extern "C" void kernel_launch(void* const* d_in, const int* in_sizes, int n_in,
                              void* d_out, int out_size, void* d_ws, size_t ws_size,
                              hipStream_t stream) {
  const float* X     = (const float*)d_in[0];
  const float* sizes = (const float*)d_in[1];
  const float* mask  = (const float*)d_in[2];
  const float* Wq    = (const float*)d_in[3];
  const float* Wk    = (const float*)d_in[4];
  const float* Wv    = (const float*)d_in[5];
  const float* Wo    = (const float*)d_in[6];
  float* out = (float*)d_out;

  const size_t ND = (size_t)B_ * N_ * D_;   // 2,097,152 elements
  bf16_t* ws  = (bf16_t*)d_ws;
  bf16_t* Xbf = ws;              ws += ND;
  bf16_t* WqT = ws;              ws += 128 * 128;
  bf16_t* WkT = ws;              ws += 128 * 128;
  bf16_t* WvT = ws;              ws += 128 * 128;
  bf16_t* WoT = ws;              ws += 128 * 128;
  bf16_t* Qbf = ws;              ws += ND;
  bf16_t* Kbf = ws;              ws += ND;
  bf16_t* Vt  = ws;              ws += ND;   // ~16.9 MB total

  k_cvt_x<<<(int)((ND + 255) / 256), 256, 0, stream>>>(X, Xbf, (int)ND);
  k_cvt_wT<<<1, 256, 0, stream>>>(Wq, WqT);
  k_cvt_wT<<<1, 256, 0, stream>>>(Wk, WkT);
  k_cvt_wT<<<1, 256, 0, stream>>>(Wv, WvT);
  k_cvt_wT<<<1, 256, 0, stream>>>(Wo, WoT);

  k_qkv<<<dim3(128, 3), 256, 0, stream>>>(Xbf, WqT, WkT, WvT, Qbf, Kbf, Vt);
  k_attn<<<128, 256, 0, stream>>>(Qbf, Kbf, Vt, sizes, mask, WoT, out);
}
